// CausalSelfAttention_84258668413287
// MI455X (gfx1250) — matrix operand order repaired and checked by reading, not compiled, not run on hardware
//
#include <hip/hip_runtime.h>
#include <stdint.h>

// Problem constants (from reference): B=2, T=2048, DIM=1024, H=16, HD=64
#define BB   2
#define TT   2048
#define DIMC 1024
#define HH   16
#define HDC  64
#define N3   3072
#define MROWS (BB * TT)   // 4096

typedef __attribute__((ext_vector_type(16))) __bf16 v16bf;
typedef __attribute__((ext_vector_type(8)))  __bf16 v8bf;
typedef __attribute__((ext_vector_type(8)))  float  v8f;

#if __has_builtin(__builtin_amdgcn_global_load_async_to_lds_b128) && \
    __has_builtin(__builtin_amdgcn_s_wait_asynccnt)
#define USE_ASYNC_LDS 1
// Builtin prototype is "vV4i*1V4i*3IiIi": (global v4i*, lds v4i*, imm off, imm cpol)
typedef int v4i_g __attribute__((vector_size(16)));
typedef __attribute__((address_space(1))) v4i_g* gptr_v4;
typedef __attribute__((address_space(3))) v4i_g* lptr_v4;
#else
#define USE_ASYNC_LDS 0
#endif

static __device__ __forceinline__ __bf16 f2bf(float f) {
  uint32_t u = __float_as_uint(f);
  uint32_t r = u + 0x7FFFu + ((u >> 16) & 1u);   // round-to-nearest-even
  uint16_t h = (uint16_t)(r >> 16);
  return __builtin_bit_cast(__bf16, h);
}

static __device__ __forceinline__ v8f wmma_bf16(v16bf a, v16bf b, v8f c) {
  return __builtin_amdgcn_wmma_f32_16x16x32_bf16(false, a, false, b, (short)0, c,
                                                 false, false);
}

#define CONCAT16(lo, hi) \
  __builtin_shufflevector((lo), (hi), 0, 1, 2, 3, 4, 5, 6, 7, 8, 9, 10, 11, 12, 13, 14, 15)

// ---------------------------------------------------------------------------
// Kernel 1: qkv = x @ W_qkv  (M=4096, N=3072, K=1024), fp32 in -> bf16 out
// scattered into Q[B,H,T,HD], K[B,H,T,HD], V^T[B,H,HD,T]
// Block tile 64x256, 8 waves x (2x4) WMMA subtiles, K-step 32.
// ---------------------------------------------------------------------------
__global__ __launch_bounds__(256)
void qkv_gemm_kernel(const float* __restrict__ X, const float* __restrict__ Wqkv,
                     __bf16* __restrict__ Qb, __bf16* __restrict__ Kb,
                     __bf16* __restrict__ Vt) {
  __shared__ __align__(16) __bf16 As[64][40];     // [m][k], 80B row stride
  __shared__ __align__(16) __bf16 Bst[256][40];   // [n][k] transposed
  const int tid  = threadIdx.x;
  const int lane = tid & 31;
  const int w    = tid >> 5;    // 0..7
  const int wm   = w & 1;       // 2 x 32 rows
  const int wn   = w >> 1;      // 4 x 64 cols
  const int bm0  = blockIdx.x * 64;
  const int bn0  = blockIdx.y * 256;

  v8f zero = {};
  v8f acc[2][4];
#pragma unroll
  for (int i = 0; i < 2; ++i)
#pragma unroll
    for (int j = 0; j < 4; ++j) acc[i][j] = zero;

  const int ar = tid >> 2;         // 0..63
  const int ac = (tid & 3) * 8;    // 0,8,16,24

  for (int k0 = 0; k0 < DIMC; k0 += 32) {
    __syncthreads();
    {  // A tile: 8 contiguous fp32 per thread -> one b128 LDS store
      const float* src = X + (size_t)(bm0 + ar) * DIMC + k0 + ac;
      v8bf ra;
#pragma unroll
      for (int j = 0; j < 8; ++j) ra[j] = f2bf(src[j]);
      *(v8bf*)&As[ar][ac] = ra;
    }
    {  // B tile: thread owns column bn0+tid; per-k reads coalesced across block
      const float* src = Wqkv + (size_t)k0 * N3 + bn0 + tid;
      v8bf r[4];
#pragma unroll
      for (int j = 0; j < 32; ++j) r[j >> 3][j & 7] = f2bf(src[(size_t)j * N3]);
#pragma unroll
      for (int c = 0; c < 4; ++c) *(v8bf*)&Bst[tid][c * 8] = r[c];
    }
    __syncthreads();

    const int hi   = (lane >= 16) ? 8 : 0;
    const int lo16 = lane & 15;
    const int kbl  = (lane < 16) ? 0 : 16;
    v16bf a[2], b[4];
#pragma unroll
    for (int mt = 0; mt < 2; ++mt) {
      const int row = wm * 32 + mt * 16 + lo16;
      v8bf l0 = *(const v8bf*)&As[row][hi];
      v8bf l1 = *(const v8bf*)&As[row][16 + hi];
      a[mt] = CONCAT16(l0, l1);
    }
#pragma unroll
    for (int nt = 0; nt < 4; ++nt) {
      const int col = wn * 64 + nt * 16 + lo16;
      v8bf b0 = *(const v8bf*)&Bst[col][hi];
      v8bf b1 = *(const v8bf*)&Bst[col][16 + hi];
      b[nt] = CONCAT16(b0, b1);
    }
#pragma unroll
    for (int mt = 0; mt < 2; ++mt)
#pragma unroll
      for (int nt = 0; nt < 4; ++nt)
        acc[mt][nt] = wmma_bf16(a[mt], b[nt], acc[mt][nt]);
  }

  // Epilogue: part/head are uniform per 16-col tile -> hoist base ptr + stride.
  const int lo16 = lane & 15;
  const int rhi  = (lane >= 16) ? 8 : 0;
#pragma unroll
  for (int mt = 0; mt < 2; ++mt) {
    const int row0 = bm0 + wm * 32 + mt * 16;
    const int bidx = row0 >> 11;   // / 2048
    const int t0   = row0 & 2047;
#pragma unroll
    for (int nt = 0; nt < 4; ++nt) {
      const int col  = bn0 + wn * 64 + nt * 16 + lo16;
      const int part = col >> 10;   // 0=q,1=k,2=v
      const int c    = col & 1023;
      const int hh   = c >> 6;
      const int dd   = c & 63;
      __bf16* bp;
      int strd;
      if (part == 0) {
        bp = Qb + (((size_t)bidx * HH + hh) * TT + t0) * HDC + dd;
        strd = HDC;
      } else if (part == 1) {
        bp = Kb + (((size_t)bidx * HH + hh) * TT + t0) * HDC + dd;
        strd = HDC;
      } else {
        bp = Vt + (((size_t)bidx * HH + hh) * HDC + dd) * TT + t0;
        strd = 1;
      }
#pragma unroll
      for (int i = 0; i < 8; ++i) bp[(i + rhi) * strd] = f2bf(acc[mt][nt][i]);
    }
  }
}

// ---------------------------------------------------------------------------
// Kernel 2: flash attention per (b,h), 16 queries per wave, key blocks of 32.
// Produces attn-out bf16 [B,T,DIM] and entropy (head-mean) via atomicAdd.
// ---------------------------------------------------------------------------
__global__ __launch_bounds__(128)
void attn_kernel(const __bf16* __restrict__ Qb, const __bf16* __restrict__ Kb,
                 const __bf16* __restrict__ Vt, __bf16* __restrict__ AO,
                 float* __restrict__ ent) {
  __shared__ __align__(16) __bf16 Pst[4][16][32];
  const int tid  = threadIdx.x;
  const int lane = tid & 31;
  const int w    = tid >> 5;          // 0..3
  const int lo16 = lane & 15;
  const int rhi  = (lane >= 16) ? 8 : 0;
  const int bh   = blockIdx.y;        // 0..31
  const int b    = bh >> 4, h = bh & 15;
  const int q0   = blockIdx.x * 64 + w * 16;

  // Q A-fragments (two K-halves of HD=64); per-lane contiguous 16B chunks.
  const __bf16* Qrow = Qb + ((size_t)bh * TT + q0 + lo16) * HDC;
  v16bf qa[2];
#pragma unroll
  for (int h2 = 0; h2 < 2; ++h2) {
    const __bf16* p = Qrow + 32 * h2 + ((lane >= 16) ? 16 : 0);
    v8bf c0 = *(const v8bf*)p;
    v8bf c1 = *(const v8bf*)(p + 8);
    qa[h2] = CONCAT16(c0, c1);
  }

  v8f zero = {};
  v8f o[4];
#pragma unroll
  for (int n4 = 0; n4 < 4; ++n4) o[n4] = zero;
  float m[8], S1[8], S2[8];
#pragma unroll
  for (int i = 0; i < 8; ++i) { m[i] = -1e30f; S1[i] = 0.0f; S2[i] = 0.0f; }

  const float scale = 0.125f;  // 1/sqrt(64)
  const int   kblo  = (lane < 16) ? 0 : 16;
  const int   nkb   = q0 / 32 + 1;   // causal: key blocks up to the diagonal

  for (int kb = 0; kb < nkb; ++kb) {
    const int k0 = kb * 32;
    float s[2][8];
#pragma unroll
    for (int nt = 0; nt < 2; ++nt) {
      const int key = k0 + nt * 16 + lo16;
      const __bf16* kp = Kb + ((size_t)bh * TT + key) * HDC + kblo;
      v8f cc = zero;
#pragma unroll
      for (int h2 = 0; h2 < 2; ++h2) {
        v16bf kf = *(const v16bf*)(kp + 32 * h2);   // contiguous 32B B-fragment
        cc = wmma_bf16(qa[h2], kf, cc);
      }
#pragma unroll
      for (int i = 0; i < 8; ++i) {
        float sv = cc[i] * scale;
        if (key > q0 + i + rhi) sv = -1e30f;        // causal mask
        s[nt][i] = sv;
      }
    }

    // Online softmax per row (rows live in 16-lane groups of the C layout).
#pragma unroll
    for (int i = 0; i < 8; ++i) {
      float rmax = fmaxf(s[0][i], s[1][i]);
#pragma unroll
      for (int xm = 8; xm >= 1; xm >>= 1)
        rmax = fmaxf(rmax, __shfl_xor(rmax, xm, 32));
      const float mn    = fmaxf(m[i], rmax);
      const float alpha = __expf(m[i] - mn);
      m[i] = mn;
      const float p0 = __expf(s[0][i] - mn);
      const float p1 = __expf(s[1][i] - mn);
      float ls1 = p0 + p1;
      float ls2 = p0 * (s[0][i] - mn) + p1 * (s[1][i] - mn);
#pragma unroll
      for (int xm = 8; xm >= 1; xm >>= 1) {
        ls1 += __shfl_xor(ls1, xm, 32);
        ls2 += __shfl_xor(ls2, xm, 32);
      }
      S1[i] = S1[i] * alpha + ls1;
      S2[i] = S2[i] * alpha + ls2;
#pragma unroll
      for (int n4 = 0; n4 < 4; ++n4) o[n4][i] *= alpha;
      // stage P (C layout -> LDS row-major) for A-fragment reload
      Pst[w][i + rhi][lo16]      = f2bf(p0);
      Pst[w][i + rhi][16 + lo16] = f2bf(p1);
    }
    // Same-wave LDS RAW: DS ops are in-order, but wait + compiler fence anyway.
    asm volatile("s_wait_dscnt 0" ::: "memory");

    v16bf pa;
    {
      const int abase = (lane >= 16) ? 16 : 0;
      v8bf p0 = *(const v8bf*)&Pst[w][lo16][abase];
      v8bf p1 = *(const v8bf*)&Pst[w][lo16][8 + abase];
      pa = CONCAT16(p0, p1);
    }
    // O += P @ V ; V^T[B,H,HD,T] makes B-fragments contiguous 32B per lane.
#pragma unroll
    for (int n4 = 0; n4 < 4; ++n4) {
      const int dim = n4 * 16 + lo16;
      const __bf16* vp = Vt + ((size_t)bh * HDC + dim) * TT + k0 + kblo;
      v16bf vb = *(const v16bf*)vp;
      o[n4] = wmma_bf16(pa, vb, o[n4]);
    }
  }

  // Epilogue: normalize, write attn-out bf16 [B,T,DIM]; entropy accumulation.
#pragma unroll
  for (int i = 0; i < 8; ++i) {
    const int   t   = q0 + i + rhi;
    const float inv = 1.0f / S1[i];
#pragma unroll
    for (int n4 = 0; n4 < 4; ++n4) {
      const int col = h * HDC + n4 * 16 + lo16;
      AO[((size_t)b * TT + t) * DIMC + col] = f2bf(o[n4][i] * inv);
    }
    if (lo16 == 0) {  // one lane per row-group (lanes 0 and 16)
      const float He = __logf(S1[i]) - S2[i] * inv;
      atomicAdd(&ent[(size_t)b * TT + t], He * (1.0f / HH));
    }
  }
}

// ---------------------------------------------------------------------------
// Kernel 3: out = attnout @ W_proj  (M=4096, N=1024, K=1024), fp32 output.
// A tile is bf16 in ws: staged via async global->LDS copies when available.
// ---------------------------------------------------------------------------
__global__ __launch_bounds__(256)
void proj_gemm_kernel(const __bf16* __restrict__ A, const float* __restrict__ Wp,
                      float* __restrict__ out) {
  __shared__ __align__(16) __bf16 As[64][40];
  __shared__ __align__(16) __bf16 Bst[256][40];   // [n][k] transposed
  const int tid  = threadIdx.x;
  const int lane = tid & 31;
  const int w    = tid >> 5;
  const int wm   = w & 1;
  const int wn   = w >> 1;
  const int bm0  = blockIdx.x * 64;
  const int bn0  = blockIdx.y * 256;

  v8f zero = {};
  v8f acc[2][4];
#pragma unroll
  for (int i = 0; i < 2; ++i)
#pragma unroll
    for (int j = 0; j < 4; ++j) acc[i][j] = zero;

  const int ar = tid >> 2;
  const int ac = (tid & 3) * 8;

  for (int k0 = 0; k0 < DIMC; k0 += 32) {
    __syncthreads();
    {  // A tile (bf16): 16B per thread, async DMA into LDS when available
      const __bf16* src = A + (size_t)(bm0 + ar) * DIMC + k0 + ac;
#if USE_ASYNC_LDS
      __builtin_amdgcn_global_load_async_to_lds_b128(
          (gptr_v4)(const void*)src, (lptr_v4)(void*)&As[ar][ac], 0, 0);
#else
      *(v8bf*)&As[ar][ac] = *(const v8bf*)src;
#endif
    }
    {  // B tile: thread owns column bn0+tid
      const float* src = Wp + (size_t)k0 * DIMC + bn0 + tid;
      v8bf r[4];
#pragma unroll
      for (int j = 0; j < 32; ++j) r[j >> 3][j & 7] = f2bf(src[(size_t)j * DIMC]);
#pragma unroll
      for (int c = 0; c < 4; ++c) *(v8bf*)&Bst[tid][c * 8] = r[c];
    }
#if USE_ASYNC_LDS
    __builtin_amdgcn_s_wait_asynccnt(0);
#endif
    __syncthreads();

    const int hi   = (lane >= 16) ? 8 : 0;
    const int lo16 = lane & 15;
    const int kbl  = (lane < 16) ? 0 : 16;
    v16bf a[2], b[4];
#pragma unroll
    for (int mt = 0; mt < 2; ++mt) {
      const int row = wm * 32 + mt * 16 + lo16;
      v8bf l0 = *(const v8bf*)&As[row][hi];
      v8bf l1 = *(const v8bf*)&As[row][16 + hi];
      a[mt] = CONCAT16(l0, l1);
    }
#pragma unroll
    for (int nt = 0; nt < 4; ++nt) {
      const int col = wn * 64 + nt * 16 + lo16;
      v8bf b0 = *(const v8bf*)&Bst[col][hi];
      v8bf b1 = *(const v8bf*)&Bst[col][16 + hi];
      b[nt] = CONCAT16(b0, b1);
    }
#pragma unroll
    for (int mt = 0; mt < 2; ++mt)
#pragma unroll
      for (int nt = 0; nt < 4; ++nt)
        acc[mt][nt] = wmma_bf16(a[mt], b[nt], acc[mt][nt]);
  }

  const int lo16 = lane & 15;
  const int rhi  = (lane >= 16) ? 8 : 0;
#pragma unroll
  for (int mt = 0; mt < 2; ++mt) {
    const int row0 = bm0 + wm * 32 + mt * 16;
#pragma unroll
    for (int nt = 0; nt < 4; ++nt) {
      const int col = bn0 + wn * 64 + nt * 16 + lo16;
      float* bp = out + (size_t)row0 * DIMC + col;
#pragma unroll
      for (int i = 0; i < 8; ++i) bp[(size_t)(i + rhi) * DIMC] = acc[mt][nt][i];
    }
  }
}

// ---------------------------------------------------------------------------
// Launch: memset entropy tail, then qkv -> attention -> proj on `stream`.
// Workspace use: 4 x 8MB bf16 buffers (Q, K, V^T, attn-out) = 32 MB.
// ---------------------------------------------------------------------------
extern "C" void kernel_launch(void* const* d_in, const int* in_sizes, int n_in,
                              void* d_out, int out_size, void* d_ws, size_t ws_size,
                              hipStream_t stream) {
  const float* X     = (const float*)d_in[0];
  const float* Wqkv  = (const float*)d_in[1];
  const float* Wproj = (const float*)d_in[2];
  float* out = (float*)d_out;
  float* ent = out + (size_t)BB * TT * DIMC;   // entropy tail of d_out

  const size_t nqkv = (size_t)BB * HH * TT * HDC;   // 4,194,304 elems each
  __bf16* Qb = (__bf16*)d_ws;
  __bf16* Kb = Qb + nqkv;
  __bf16* Vt = Kb + nqkv;
  __bf16* AO = Vt + nqkv;

  (void)hipMemsetAsync(ent, 0, (size_t)BB * TT * sizeof(float), stream);

  dim3 g1(MROWS / 64, N3 / 256);      // 64 x 12
  qkv_gemm_kernel<<<g1, 256, 0, stream>>>(X, Wqkv, Qb, Kb, Vt);

  dim3 g2(TT / 64, BB * HH);          // 32 x 32
  attn_kernel<<<g2, 128, 0, stream>>>(Qb, Kb, Vt, AO, ent);

  dim3 g3(MROWS / 64, DIMC / 256);    // 64 x 4
  proj_gemm_kernel<<<g3, 256, 0, stream>>>(AO, Wproj, out);
}
